// VigCGENNClassifier_29927332118824
// MI455X (gfx1250) — matrix operand order repaired and checked by reading, not compiled, not run on hardware
//
#include <hip/hip_runtime.h>

// ---------------- problem constants (from reference) ----------------
#define EMB     192
#define HID     64
#define LAYERS  6
#define IMG     224
#define FMAP    56
#define NPATCH  (FMAP*FMAP)        // 3136
#define BATCH   8
#define BN      (BATCH*NPATCH)     // 25088
#define KNN     8
#define NEDGE   (BN*KNN)           // 200704

typedef __attribute__((ext_vector_type(16))) __bf16 v16bf;
typedef __attribute__((ext_vector_type(8)))  float  v8f;

__device__ __forceinline__ float sigmoidf_(float x) { return 1.0f / (1.0f + __expf(-x)); }

// =====================================================================
// WMMA GEMM, K known at compile time (64 / 128 / 192):
//   C[M,N] = A[M,K] * Bt[N,K]^T (+ bias[N])
// A, Bt fp32 in HBM, converted to bf16; fp32 accumulate via
// v_wmma_f32_16x16x32_bf16.  Bt has a per-element K stride so per-grade
// weight slices W[:, :, g] are consumed in place.
// One wave owns a 16x32 D tile (one A fragment feeds two WMMAs -> halves
// global A traffic per FLOP); 4 waves/block share a 32-col B tile in LDS.
// K-loop fully unrolled: 2*K/32 back-to-back WMMAs per wave.
// =====================================================================
template<int K>
__global__ __launch_bounds__(128)
void wmma_gemm_bf16(const float* __restrict__ A, int lda,
                    const float* __restrict__ Bt, int ldb, int kstride,
                    const float* __restrict__ bias,
                    float* __restrict__ C, int ldc,
                    int M, int N)
{
    __shared__ __bf16 sB[32 * K];
    const int lane = threadIdx.x;             // 0..31
    const int wave = threadIdx.y;             // 0..3
    const int n0   = blockIdx.y * 32;
    const int m0   = (blockIdx.x * 4 + wave) * 16;

    // cooperatively stage the 32-column weight tile (bf16) into LDS
    const int tid = wave * 32 + lane;
    for (int idx = tid; idx < 32 * K; idx += 128) {
        int c = idx / K, kk = idx - c * K;
        sB[idx] = (__bf16)Bt[(size_t)(n0 + c) * ldb + (size_t)kk * kstride];
    }
    __syncthreads();

    v8f acc0 = {0.f, 0.f, 0.f, 0.f, 0.f, 0.f, 0.f, 0.f};
    v8f acc1 = {0.f, 0.f, 0.f, 0.f, 0.f, 0.f, 0.f, 0.f};
    if (m0 < M) {                             // wave-uniform: EXEC stays all-ones
        const int r  = lane & 15;
        const int hi = lane >> 4;
        const float*  arow  = A + (size_t)(m0 + r) * lda;
        const __bf16* bcol0 = sB + (size_t)(lane & 15) * K;
        const __bf16* bcol1 = sB + (size_t)(16 + (lane & 15)) * K;
#pragma unroll
        for (int k0 = 0; k0 < K; k0 += 32) {
            if (k0 + 32 < K) __builtin_prefetch(arow + k0 + 32, 0, 1); // global_prefetch_b8
            v16bf a, b0, b1;
#pragma unroll
            for (int v = 0; v < 8; ++v) {
                // ISA 7.12.2: 16-bit A 16x32 — lanes0-15 K=0..7,16..23; lanes16-31 K=8..15,24..31
                int ka = k0 + ((v < 4) ? (8 * hi + 2 * v) : (16 + 8 * hi + 2 * (v - 4)));
                a[2 * v]     = (__bf16)arow[ka];
                a[2 * v + 1] = (__bf16)arow[ka + 1];
                // B 32x16 — same K order as A: lanes0-15 K=0..7,16..23; lanes16-31 K=8..15,24..31
                int kb = ka;
                b0[2 * v]     = bcol0[kb];
                b0[2 * v + 1] = bcol0[kb + 1];
                b1[2 * v]     = bcol1[kb];
                b1[2 * v + 1] = bcol1[kb + 1];
            }
            acc0 = __builtin_amdgcn_wmma_f32_16x16x32_bf16(
                       false, a, false, b0, (short)0, acc0, false, false);
            acc1 = __builtin_amdgcn_wmma_f32_16x16x32_bf16(
                       false, a, false, b1, (short)0, acc1, false, false);
        }
        const int col0 = n0 + (lane & 15);
        const int col1 = col0 + 16;
        float bv0 = (bias != nullptr) ? bias[col0] : 0.0f;
        float bv1 = (bias != nullptr && col1 < N) ? bias[col1] : 0.0f;
#pragma unroll
        for (int j = 0; j < 8; ++j) {         // D: row = m0 + 8*hi + j, col = lane&15
            int row = m0 + hi * 8 + j;
            if (row < M) {
                C[(size_t)row * ldc + col0] = acc0[j] + bv0;
                if (col1 < N) C[(size_t)row * ldc + col1] = acc1[j] + bv1;
            }
        }
    }
}

// =====================================================================
// Stem: 4x4/s4 conv, per-sample LayerNorm over (C,H,W), pos-embed.
// Output layout feat[b*N + n][c]  (channel-last, GEMM ready).
// =====================================================================
__global__ __launch_bounds__(192)
void stem_conv(const float* __restrict__ x, const float* __restrict__ w,
               const float* __restrict__ bias, float* __restrict__ out)
{
    __shared__ float tile[48];
    int blk = blockIdx.x;                 // 0..BN-1
    int b = blk / NPATCH, n = blk - b * NPATCH;
    int py = n / FMAP, px = n - py * FMAP;
    int c = threadIdx.x;
    if (c < 48) {
        int ci = c >> 4, rr = c & 15, ky = rr >> 2, kx = rr & 3;
        tile[c] = x[(((size_t)b * 3 + ci) * IMG + (py * 4 + ky)) * IMG + (px * 4 + kx)];
    }
    __syncthreads();
    float acc = bias[c];
#pragma unroll
    for (int t = 0; t < 48; ++t) acc += tile[t] * w[c * 48 + t];
    out[(size_t)blk * EMB + c] = acc;
}

__global__ __launch_bounds__(256)
void stem_stats(const float* __restrict__ feat, float* __restrict__ stats)
{
    __shared__ float ssum[256], ssq[256];
    int b = blockIdx.x;
    const size_t n = (size_t)NPATCH * EMB;
    const float* p = feat + (size_t)b * n;
    float s = 0.f, q = 0.f;
    for (size_t i = threadIdx.x; i < n; i += 256) { float v = p[i]; s += v; q += v * v; }
    ssum[threadIdx.x] = s; ssq[threadIdx.x] = q;
    __syncthreads();
    for (int st = 128; st > 0; st >>= 1) {
        if ((int)threadIdx.x < st) { ssum[threadIdx.x] += ssum[threadIdx.x + st];
                                     ssq[threadIdx.x]  += ssq[threadIdx.x + st]; }
        __syncthreads();
    }
    if (threadIdx.x == 0) {
        float mu = ssum[0] / (float)n;
        stats[2 * b]     = mu;
        stats[2 * b + 1] = ssq[0] / (float)n - mu * mu;
    }
}

__global__ void stem_norm(float* __restrict__ feat, const float* __restrict__ stats,
                          const float* __restrict__ g, const float* __restrict__ bta,
                          const float* __restrict__ pos)
{
    size_t t = (size_t)blockIdx.x * blockDim.x + threadIdx.x;
    if (t >= (size_t)BN * EMB) return;
    size_t per = (size_t)NPATCH * EMB;
    int b = (int)(t / per);
    size_t r = t - (size_t)b * per;
    int n = (int)(r / EMB), c = (int)(r - (size_t)n * EMB);
    float mu = stats[2 * b], var = stats[2 * b + 1];
    size_t gi = (size_t)c * NPATCH + n;
    feat[t] = (feat[t] - mu) * rsqrtf(var + 1e-5f) * g[gi] + bta[gi] + pos[gi];
}

// blades 1/2 of the embed are rank-1 (only the 193rd input is nonzero there)
__global__ void embed_vec(float* __restrict__ hcat, const float* __restrict__ inW,
                          const float* __restrict__ coords)
{
    int t = blockIdx.x * blockDim.x + threadIdx.x;
    if (t >= BN * HID) return;
    int v = t / HID, c = t - v * HID;
    int n = v % NPATCH;
    float wl = inW[c * 193 + 192];
    size_t base = (size_t)v * 128 + c;
    size_t ps = (size_t)BN * 128;
    hcat[ps * 1 + base] = wl * coords[2 * n + 0];
    hcat[ps * 2 + base] = wl * coords[2 * n + 1];
    hcat[ps * 3 + base] = 0.0f;
}

// =====================================================================
// sgp elementwise pieces
// =====================================================================
__global__ void normalize_mv_k(float* __restrict__ xr, const float* __restrict__ na)
{
    int t = blockIdx.x * blockDim.x + threadIdx.x;
    if (t >= BN * HID) return;
    int c = t % HID;
    size_t ps = (size_t)BN * HID;
    float x0 = xr[t], x1 = xr[ps + t], x2 = xr[2 * ps + t], x3 = xr[3 * ps + t];
    float n0 = sqrtf(x0 * x0 + 1e-12f);
    float n1 = sqrtf(x1 * x1 + x2 * x2 + 1e-12f);
    float n2 = sqrtf(x3 * x3 + 1e-12f);
    float d0 = sigmoidf_(na[c * 3 + 0]) * (n0 - 1.f) + 1.f + 1e-6f;
    float d1 = sigmoidf_(na[c * 3 + 1]) * (n1 - 1.f) + 1.f + 1e-6f;
    float d2 = sigmoidf_(na[c * 3 + 2]) * (n2 - 1.f) + 1.f + 1e-6f;
    xr[t] = x0 / d0; xr[ps + t] = x1 / d1; xr[2 * ps + t] = x2 / d1; xr[3 * ps + t] = x3 / d2;
}

// Cayley x path-weight contraction unrolled to its 16 nonzero terms.
// Path index table (gi,gj,gk): 0:(0,0,0) 1:(0,1,1) 2:(0,2,2) 3:(1,0,1) 4:(1,1,0)
//                              5:(1,1,2) 6:(1,2,1) 7:(2,0,2) 8:(2,1,1) 9:(2,2,0)
// mode 0: write sgp result; 1: h = silu(sgp); 2: h += silu(sgp)
__global__ void sgp_combine(const float* __restrict__ x, size_t xps, int xlda,
                            const float* __restrict__ xr,
                            const float* __restrict__ tl,
                            const float* __restrict__ gw,
                            float* __restrict__ hout, size_t hps, int hlda,
                            const float* __restrict__ aa, const float* __restrict__ ab,
                            int mode)
{
    int t = blockIdx.x * blockDim.x + threadIdx.x;
    if (t >= BN * HID) return;
    int v = t / HID, c = t - v * HID;
    size_t xb = (size_t)v * xlda + c;
    float x0 = x[xb], x1 = x[xps + xb], x2 = x[2 * xps + xb], x3 = x[3 * xps + xb];
    size_t ps = (size_t)BN * HID;
    float r0 = xr[t], r1 = xr[ps + t], r2 = xr[2 * ps + t], r3 = xr[3 * ps + t];
    const float* g = gw + c * 10;
    float gp0 = g[0] * x0 * r0 + g[3] * (x1 * r1 + x2 * r2) + g[7] * x3 * r3;
    float gp1 = g[1] * x0 * r1 + g[4] * x1 * r0 - g[5] * x2 * r3 - g[8] * x3 * r2;
    float gp2 = g[1] * x0 * r2 + g[5] * x1 * r3 + g[4] * x2 * r0 + g[8] * x3 * r1;
    float gp3 = g[2] * x0 * r3 + g[6] * x1 * r2 - g[6] * x2 * r1 - g[9] * x3 * r0;
    const float is2 = 0.70710678118654752440f;
    float y0 = (tl[t] + gp0) * is2;
    float y1 = (tl[ps + t] + gp1) * is2;
    float y2 = (tl[2 * ps + t] + gp2) * is2;
    float y3 = (tl[3 * ps + t] + gp3) * is2;
    if (mode != 0) {
        float i0 = y0, i1 = y1 * y1 + y2 * y2, i2 = y3 * y3;
        float s0 = sigmoidf_(aa[c * 3 + 0] * i0 + ab[c * 3 + 0]);
        float s1 = sigmoidf_(aa[c * 3 + 1] * i1 + ab[c * 3 + 1]);
        float s2 = sigmoidf_(aa[c * 3 + 2] * i2 + ab[c * 3 + 2]);
        y0 *= s0; y1 *= s1; y2 *= s1; y3 *= s2;
    }
    size_t hb = (size_t)v * hlda + c;
    if (mode == 2) {
        y0 += hout[hb]; y1 += hout[hps + hb]; y2 += hout[2 * hps + hb]; y3 += hout[3 * hps + hb];
    }
    hout[hb] = y0; hout[hps + hb] = y1; hout[2 * hps + hb] = y2; hout[3 * hps + hb] = y3;
}

// =====================================================================
// Graph layer pieces
// =====================================================================
__global__ void edge_diff(const float* __restrict__ hplane,
                          const int* __restrict__ rows, const int* __restrict__ cols,
                          float* __restrict__ diff)
{
    size_t t = (size_t)blockIdx.x * blockDim.x + threadIdx.x;
    if (t >= (size_t)NEDGE * HID) return;
    size_t e = t / HID; int c = (int)(t - e * HID);
    int rv = rows[e], cv = cols[e];
    diff[t] = hplane[(size_t)rv * 128 + c] - hplane[(size_t)cv * 128 + c];
}

// rows = repeat(arange(BN), 8): node v's edges are contiguous [v*8, v*8+8)
// -> fuse mv_silu gating with segment_sum, write into agg half of hcat.
__global__ void gate_agg(const float* __restrict__ m,
                         const float* __restrict__ sa, const float* __restrict__ sb,
                         float* __restrict__ hcat)
{
    int t = blockIdx.x * blockDim.x + threadIdx.x;
    if (t >= BN * HID) return;
    int v = t / HID, c = t - v * HID;
    size_t ps = (size_t)NEDGE * HID;
    float a0 = sa[c * 3 + 0], a1 = sa[c * 3 + 1], a2 = sa[c * 3 + 2];
    float b0 = sb[c * 3 + 0], b1 = sb[c * 3 + 1], b2 = sb[c * 3 + 2];
    float c0 = 0.f, c1 = 0.f, c2 = 0.f, c3 = 0.f;
#pragma unroll
    for (int j = 0; j < KNN; ++j) {
        size_t e = ((size_t)v * KNN + j) * HID + c;
        float m0 = m[e], m1 = m[ps + e], m2 = m[2 * ps + e], m3 = m[3 * ps + e];
        float i1 = m1 * m1 + m2 * m2, i2 = m3 * m3;
        float s0 = sigmoidf_(a0 * m0 + b0);
        float s1 = sigmoidf_(a1 * i1 + b1);
        float s2 = sigmoidf_(a2 * i2 + b2);
        c0 += s0 * m0; c1 += s1 * m1; c2 += s1 * m2; c3 += s2 * m3;
    }
    size_t hps = (size_t)BN * 128;
    size_t hb = (size_t)v * 128 + 64 + c;
    hcat[hb] = c0; hcat[hps + hb] = c1; hcat[2 * hps + hb] = c2; hcat[3 * hps + hb] = c3;
}

// =====================================================================
// Readout
// =====================================================================
__global__ __launch_bounds__(192)
void out_ln(float* __restrict__ s, const float* __restrict__ g, const float* __restrict__ b)
{
    __shared__ float red[256], red2[256];
    int v = blockIdx.x, c = threadIdx.x;
    float val = s[(size_t)v * EMB + c];
    red[c] = val; red2[c] = val * val;
    if (c < 64) { red[c + 192] = 0.f; red2[c + 192] = 0.f; }
    __syncthreads();
    for (int st = 128; st > 0; st >>= 1) {
        if (c < st) { red[c] += red[c + st]; red2[c] += red2[c + st]; }
        __syncthreads();
    }
    float mu = red[0] / (float)EMB;
    float var = red2[0] / (float)EMB - mu * mu;
    s[(size_t)v * EMB + c] = (val - mu) * rsqrtf(var + 1e-5f) * g[c] + b[c];
}

__global__ __launch_bounds__(256)
void pool_k(const float* __restrict__ s, float* __restrict__ pooled)
{
    int b = blockIdx.x, c = blockIdx.y;
    __shared__ float red[256];
    float acc = 0.f;
    for (int n = threadIdx.x; n < NPATCH; n += 256)
        acc += s[((size_t)b * NPATCH + n) * EMB + c];
    red[threadIdx.x] = acc;
    __syncthreads();
    for (int st = 128; st > 0; st >>= 1) {
        if ((int)threadIdx.x < st) red[threadIdx.x] += red[threadIdx.x + st];
        __syncthreads();
    }
    if (threadIdx.x == 0) pooled[b * EMB + c] = red[0] / (float)NPATCH;
}

__global__ void classify(const float* __restrict__ pooled, const float* __restrict__ W,
                         const float* __restrict__ b, float* __restrict__ out)
{
    int t = blockIdx.x * blockDim.x + threadIdx.x;
    if (t >= BATCH * 10) return;
    int bb = t / 10, k = t - bb * 10;
    float acc = b[k];
    for (int c = 0; c < EMB; ++c) acc += pooled[bb * EMB + c] * W[k * EMB + c];
    out[t] = acc;
}

// =====================================================================
// Host orchestration
// =====================================================================
extern "C" void kernel_launch(void* const* d_in, const int* in_sizes, int n_in,
                              void* d_out, int out_size, void* d_ws, size_t ws_size,
                              hipStream_t stream)
{
    (void)in_sizes; (void)n_in; (void)out_size; (void)ws_size;
    const float* x      = (const float*)d_in[0];
    const float* conv_w = (const float*)d_in[1];
    const float* conv_b = (const float*)d_in[2];
    const float* stem_g = (const float*)d_in[3];
    const float* stem_b = (const float*)d_in[4];
    const float* pos    = (const float*)d_in[5];
    const float* coords = (const float*)d_in[6];
    const float* in_W   = (const float*)d_in[7];
    const float* in_b   = (const float*)d_in[8];
    const float* in_Wl  = (const float*)d_in[9];
    const float* in_bl  = (const float*)d_in[10];
    const float* in_Wr  = (const float*)d_in[11];
    const float* in_na  = (const float*)d_in[12];
    const float* in_gw  = (const float*)d_in[13];
    const float* eg_eW  = (const float*)d_in[14];
    const float* eg_eb  = (const float*)d_in[15];
    const float* eg_sa  = (const float*)d_in[16];
    const float* eg_sb  = (const float*)d_in[17];
    const float* eg_nW  = (const float*)d_in[18];
    const float* eg_nb  = (const float*)d_in[19];
    const float* eg_Wl  = (const float*)d_in[20];
    const float* eg_bl  = (const float*)d_in[21];
    const float* eg_Wr  = (const float*)d_in[22];
    const float* eg_na  = (const float*)d_in[23];
    const float* eg_gw  = (const float*)d_in[24];
    const float* eg_aa  = (const float*)d_in[25];
    const float* eg_ab  = (const float*)d_in[26];
    const float* out_W  = (const float*)d_in[27];
    const float* out_b  = (const float*)d_in[28];
    const float* on_g   = (const float*)d_in[29];
    const float* on_b   = (const float*)d_in[30];
    const float* cls_W  = (const float*)d_in[31];
    const float* cls_b  = (const float*)d_in[32];
    const int*   rows   = (const int*)d_in[33];
    const int*   cols   = (const int*)d_in[34];

    // ---- workspace partition (floats) ----
    float* ws = (float*)d_ws;
    size_t off = 0;
    auto alloc = [&](size_t n) { float* p = ws + off; off += n; return p; };
    const size_t psH = (size_t)BN * HID;       // 64-wide plane stride
    const size_t psC = (size_t)BN * 128;       // hcat plane stride
    const size_t psE = (size_t)NEDGE * HID;    // edge plane stride
    float* feat   = alloc((size_t)BN * EMB);
    float* stats  = alloc(2 * BATCH);
    float* hcat   = alloc(4 * psC);            // [blade][node][ h(64) | agg(64) ]
    float* xr     = alloc(4 * psH);
    float* tl     = alloc(4 * psH);
    float* nu     = alloc(4 * psH);
    float* diff   = alloc(psE);                // reused per blade
    float* medge  = alloc(4 * psE);
    float* sbuf   = alloc((size_t)BN * EMB);
    float* pooled = alloc((size_t)BATCH * EMB);

    auto gemm = [&](const float* A, int lda, const float* Bt, int ldb, int kst,
                    const float* bias, float* C, int ldc, int M, int N, int K) {
        dim3 grid((M + 63) / 64, N / 32), block(32, 4);
        if (K == 64)
            wmma_gemm_bf16<64><<<grid, block, 0, stream>>>(A, lda, Bt, ldb, kst, bias, C, ldc, M, N);
        else if (K == 128)
            wmma_gemm_bf16<128><<<grid, block, 0, stream>>>(A, lda, Bt, ldb, kst, bias, C, ldc, M, N);
        else
            wmma_gemm_bf16<192><<<grid, block, 0, stream>>>(A, lda, Bt, ldb, kst, bias, C, ldc, M, N);
    };

    const int GBg[4] = {0, 1, 1, 2};
    const int TPB = 256, NB_H = (BN * HID + TPB - 1) / TPB;

    auto run_sgp = [&](const float* X, size_t xps, int xlda,
                       const float* Wl, const float* bl,
                       const float* Wr, const float* na, const float* gw,
                       const float* aa, const float* ab, int mode) {
        for (int i = 0; i < 4; ++i) {
            gemm(X + i * xps, xlda, Wr + GBg[i], HID * 3, 3, nullptr,
                 xr + i * psH, HID, BN, HID, HID);
            gemm(X + i * xps, xlda, Wl + GBg[i], HID * 3, 3, (i == 0) ? bl : nullptr,
                 tl + i * psH, HID, BN, HID, HID);
        }
        normalize_mv_k<<<NB_H, TPB, 0, stream>>>(xr, na);
        sgp_combine<<<NB_H, TPB, 0, stream>>>(X, xps, xlda, xr, tl, gw,
                                              hcat, psC, 128, aa, ab, mode);
    };

    // ---- stem ----
    stem_conv<<<BN, 192, 0, stream>>>(x, conv_w, conv_b, feat);
    stem_stats<<<BATCH, 256, 0, stream>>>(feat, stats);
    stem_norm<<<(int)(((size_t)BN * EMB + TPB - 1) / TPB), TPB, 0, stream>>>(
        feat, stats, stem_g, stem_b, pos);

    // ---- multivector embed: blade0 GEMM (K=192) + rank-1 blades 1/2 ----
    gemm(feat, EMB, in_W, 193, 1, in_b, hcat, 128, BN, HID, EMB);
    embed_vec<<<NB_H, TPB, 0, stream>>>(hcat, in_W, coords);

    // ---- input sgp (writes h half of hcat in place) ----
    run_sgp(hcat, psC, 128, in_Wl, in_bl, in_Wr, in_na, in_gw, nullptr, nullptr, 0);

    // ---- graph layers ----
    const int NB_E = (int)(((size_t)NEDGE * HID + TPB - 1) / TPB);
    for (int l = 0; l < LAYERS; ++l) {
        const float* eW = eg_eW + (size_t)l * HID * HID * 3;
        for (int i = 0; i < 4; ++i) {
            edge_diff<<<NB_E, TPB, 0, stream>>>(hcat + i * psC, rows, cols, diff);
            gemm(diff, HID, eW + GBg[i], HID * 3, 3, (i == 0) ? eg_eb + l * HID : nullptr,
                 medge + i * psE, HID, NEDGE, HID, HID);
        }
        gate_agg<<<NB_H, TPB, 0, stream>>>(medge, eg_sa + l * HID * 3, eg_sb + l * HID * 3, hcat);

        const float* nW = eg_nW + (size_t)l * HID * 128 * 3;
        for (int i = 0; i < 4; ++i)
            gemm(hcat + i * psC, 128, nW + GBg[i], 128 * 3, 3,
                 (i == 0) ? eg_nb + l * HID : nullptr, nu + i * psH, HID, BN, HID, 128);

        run_sgp(nu, psH, HID,
                eg_Wl + (size_t)l * HID * HID * 3, eg_bl + l * HID,
                eg_Wr + (size_t)l * HID * HID * 3, eg_na + l * HID * 3,
                eg_gw + (size_t)l * HID * 10,
                eg_aa + l * HID * 3, eg_ab + l * HID * 3,
                (l == 0) ? 1 : 2);
    }

    // ---- readout: blade0 projection, LN, pool, classifier ----
    gemm(hcat, 128, out_W, HID, 1, out_b, sbuf, EMB, BN, EMB, HID);
    out_ln<<<BN, 192, 0, stream>>>(sbuf, on_g, on_b);
    pool_k<<<dim3(BATCH, EMB), 256, 0, stream>>>(sbuf, pooled);
    classify<<<1, 128, 0, stream>>>(pooled, cls_W, cls_b, (float*)d_out);
}
